// TemporalAttentiveStatsPooling_47545287967313
// MI455X (gfx1250) — matrix operand order repaired and checked by reading, not compiled, not run on hardware
//
#include <hip/hip_runtime.h>
#include <hip/hip_bf16.h>
#include <math.h>

// Problem constants (from the reference)
#define BB 64
#define TT 2048
#define DD 512
#define AA 128

typedef __attribute__((ext_vector_type(16))) __bf16 v16bf;
typedef __attribute__((ext_vector_type(4)))  __bf16 v4bf;
typedef __attribute__((ext_vector_type(8)))  float  v8f;

// Branchless fast tanh: 1 - 2/(exp(2x)+1).
// exp overflow -> +inf -> result 1.0 ; exp underflow -> 0 -> result -1.0.
__device__ __forceinline__ float fast_tanh(float v)
{
    float e = __expf(2.0f * v);
    return 1.0f - 2.0f * __builtin_amdgcn_rcpf(e + 1.0f);
}

// ---------------------------------------------------------------------------
// Kernel 1: scores[b*T + t] = tanh(x[row,:] @ W1 + b1) @ W2 + b2
// One wave handles a 16-row tile x all A=128 columns (8 WMMA N-tiles).
// K loop: D=512 in steps of 32 (bf16 WMMA K); W1 slab double-buffered in LDS
// as bf16 (one barrier per K-step). A 32-row slab of W1 is contiguous memory,
// so staging is a flat float4 -> packed-4xbf16 copy-convert.
// ---------------------------------------------------------------------------
__global__ __launch_bounds__(256)
void tasp_score_kernel(const float* __restrict__ x,
                       const float* __restrict__ W1,
                       const float* __restrict__ b1,
                       const float* __restrict__ W2,
                       const float* __restrict__ b2,
                       float* __restrict__ scores)
{
    __shared__ __bf16 sW1[2][32 * AA];   // two K-slabs of W1, bf16: 16 KB

    const int tid   = threadIdx.x;
    const int wave  = tid >> 5;       // 0..7
    const int lane  = tid & 31;
    const int lhalf = lane >> 4;      // 0 or 1 (lane group)
    const int lmod  = lane & 15;

    const int rowBase = (blockIdx.x * 8 + wave) * 16;   // 16 t-rows per wave

    v8f acc[8];
    #pragma unroll
    for (int n = 0; n < 8; ++n) acc[n] = (v8f){0.f,0.f,0.f,0.f,0.f,0.f,0.f,0.f};

    // Per-lane constants for the epilogue (column a = n*16 + lmod)
    float b1v[8], w2v[8];
    #pragma unroll
    for (int n = 0; n < 8; ++n) {
        b1v[n] = b1[n * 16 + lmod];
        w2v[n] = W2[n * 16 + lmod];
    }
    const float bias2 = b2[0];

    const float* xrowBase = x + (size_t)(rowBase + lmod) * DD;

    // Vectorized slab staging: slab (32 rows x 128 cols) is contiguous.
    // float4 load -> packed 4x bf16 -> ds_store_b64 ; 4 iters/thread.
    auto stage = [&](int buf, int kb) {
        const float4* src = (const float4*)(W1 + (size_t)kb * AA);
        v4bf*         dst = (v4bf*)sW1[buf];
        #pragma unroll
        for (int i = 0; i < (32 * AA) / (256 * 4); ++i) {
            const float4 wv = src[tid + i * 256];
            v4bf pk;
            pk[0] = (__bf16)wv.x; pk[1] = (__bf16)wv.y;
            pk[2] = (__bf16)wv.z; pk[3] = (__bf16)wv.w;
            dst[tid + i * 256] = pk;
        }
    };

    stage(0, 0);
    __syncthreads();

    for (int kb = 0; kb < DD; kb += 32) {
        const int buf = (kb >> 5) & 1;

        // Stage next slab into the other buffer (overlaps with compute below)
        if (kb + 32 < DD) stage(buf ^ 1, kb + 32);

        // A fragment (16x32 bf16): lane holds row m = lmod with
        // K = {kbase+0..7, kbase+16..23}, kbase = 8*lhalf (ISA 16-bit A layout)
        // Two 8-float groups, each 16B-aligned -> 4x global_load_b128.
        const float4* xr4 = (const float4*)(xrowBase + kb + 8 * lhalf);
        if (kb + 32 < DD) {
            __builtin_prefetch((const float*)xr4 + 32, 0, 3);  // next K-chunk
        }
        const float4 a0 = xr4[0];   // K = kbase+0..3
        const float4 a1 = xr4[1];   // K = kbase+4..7
        const float4 a2 = xr4[4];   // K = kbase+16..19
        const float4 a3 = xr4[5];   // K = kbase+20..23
        v16bf afrag;
        afrag[0]  = (__bf16)a0.x; afrag[1]  = (__bf16)a0.y;
        afrag[2]  = (__bf16)a0.z; afrag[3]  = (__bf16)a0.w;
        afrag[4]  = (__bf16)a1.x; afrag[5]  = (__bf16)a1.y;
        afrag[6]  = (__bf16)a1.z; afrag[7]  = (__bf16)a1.w;
        afrag[8]  = (__bf16)a2.x; afrag[9]  = (__bf16)a2.y;
        afrag[10] = (__bf16)a2.z; afrag[11] = (__bf16)a2.w;
        afrag[12] = (__bf16)a3.x; afrag[13] = (__bf16)a3.y;
        afrag[14] = (__bf16)a3.z; afrag[15] = (__bf16)a3.w;

        #pragma unroll
        for (int n = 0; n < 8; ++n) {
            // B fragment (32x16 bf16): lane holds column N = lmod,
            // K = {8*lhalf+0..7, 8*lhalf+16..23} (same order as A fragment)
            v16bf bfrag;
            const __bf16* wc = sW1[buf] + (size_t)(8 * lhalf) * AA + n * 16 + lmod;
            #pragma unroll
            for (int e = 0; e < 16; ++e) bfrag[e] = wc[(size_t)((e & 7) + ((e >> 3) << 4)) * AA];

            acc[n] = __builtin_amdgcn_wmma_f32_16x16x32_bf16(
                false, afrag, false, bfrag, (short)0, acc[n], false, false);
        }

        // One barrier per step: staging of buf^1 done, compute on buf done.
        __syncthreads();
    }

    // Epilogue: per accumulator VGPR r, row M = r + 8*lhalf, column N = lmod.
    // score(row) = sum_a tanh(h[row,a] + b1[a]) * W2[a]
    #pragma unroll
    for (int r = 0; r < 8; ++r) {
        float p = 0.f;
        #pragma unroll
        for (int n = 0; n < 8; ++n) {
            float hv = fast_tanh(acc[n][r] + b1v[n]);
            p += hv * w2v[n];
        }
        // reduce the 16 column-partials across the 16-lane half
        #pragma unroll
        for (int off = 8; off >= 1; off >>= 1)
            p += __shfl_xor(p, off, 16);
        if (lmod == 0) {
            int row = rowBase + r + 8 * lhalf;
            scores[row] = p + bias2;
        }
    }
}

// ---------------------------------------------------------------------------
// Kernel 2: softmax over T per batch. 64 blocks x 256 threads, 8 scores/thread.
// ---------------------------------------------------------------------------
__global__ __launch_bounds__(256)
void tasp_softmax_kernel(const float* __restrict__ scores,
                         float* __restrict__ weights)
{
    __shared__ float red[256];
    const int b   = blockIdx.x;
    const int tid = threadIdx.x;
    const float* s = scores + (size_t)b * TT;
    float*       w = weights + (size_t)b * TT;

    float v[8];
    float m = -INFINITY;
    #pragma unroll
    for (int i = 0; i < 8; ++i) { v[i] = s[tid + i * 256]; m = fmaxf(m, v[i]); }
    red[tid] = m; __syncthreads();
    for (int off = 128; off > 0; off >>= 1) {
        if (tid < off) red[tid] = fmaxf(red[tid], red[tid + off]);
        __syncthreads();
    }
    m = red[0]; __syncthreads();

    float sum = 0.f;
    #pragma unroll
    for (int i = 0; i < 8; ++i) { v[i] = __expf(v[i] - m); sum += v[i]; }
    red[tid] = sum; __syncthreads();
    for (int off = 128; off > 0; off >>= 1) {
        if (tid < off) red[tid] += red[tid + off];
        __syncthreads();
    }
    const float inv = __builtin_amdgcn_rcpf(red[0]);
    #pragma unroll
    for (int i = 0; i < 8; ++i) w[tid + i * 256] = v[i] * inv;
}

// ---------------------------------------------------------------------------
// Kernel 3: weighted mu and E[x^2]. Grid (B, 16): each block sweeps T/16
// timesteps over all D=512 (128 threads x float4), accumulates locally, then
// atomically adds into acc[b][0..511]=mu, acc[b][512..1023]=m2.
// ---------------------------------------------------------------------------
__global__ __launch_bounds__(128)
void tasp_stats_kernel(const float* __restrict__ x,
                       const float* __restrict__ weights,
                       float* __restrict__ acc)
{
    const int b     = blockIdx.x;
    const int chunk = blockIdx.y;           // 0..15
    const int c     = threadIdx.x;          // float4 column 0..127
    const float4* xb = (const float4*)(x + (size_t)b * TT * DD);
    const float*  w  = weights + (size_t)b * TT;

    float4 mu = {0.f,0.f,0.f,0.f};
    float4 m2 = {0.f,0.f,0.f,0.f};
    const int t0 = chunk * (TT / 16);
    for (int t = t0; t < t0 + TT / 16; ++t) {
        const float wt = w[t];
        const float4 xv = xb[(size_t)t * (DD / 4) + c];
        mu.x += wt * xv.x;  m2.x += wt * xv.x * xv.x;
        mu.y += wt * xv.y;  m2.y += wt * xv.y * xv.y;
        mu.z += wt * xv.z;  m2.z += wt * xv.z * xv.z;
        mu.w += wt * xv.w;  m2.w += wt * xv.w * xv.w;
    }
    float* ab = acc + (size_t)b * 2 * DD;
    atomicAdd(&ab[c * 4 + 0], mu.x);
    atomicAdd(&ab[c * 4 + 1], mu.y);
    atomicAdd(&ab[c * 4 + 2], mu.z);
    atomicAdd(&ab[c * 4 + 3], mu.w);
    atomicAdd(&ab[DD + c * 4 + 0], m2.x);
    atomicAdd(&ab[DD + c * 4 + 1], m2.y);
    atomicAdd(&ab[DD + c * 4 + 2], m2.z);
    atomicAdd(&ab[DD + c * 4 + 3], m2.w);
}

// ---------------------------------------------------------------------------
// Kernel 4: finalize. out[b, d] = mu, out[b, D+d] = sqrt(E[x^2]-mu^2 + 1e-6)
// ---------------------------------------------------------------------------
__global__ __launch_bounds__(512)
void tasp_finalize_kernel(const float* __restrict__ acc,
                          float* __restrict__ out)
{
    const int b = blockIdx.x;
    const int d = threadIdx.x;
    const float mu = acc[(size_t)b * 2 * DD + d];
    const float m2 = acc[(size_t)b * 2 * DD + DD + d];
    const float var = fmaxf(m2 - mu * mu, 0.f);
    out[(size_t)b * 2 * DD + d]      = mu;
    out[(size_t)b * 2 * DD + DD + d] = sqrtf(var + 1e-6f);
}

// ---------------------------------------------------------------------------
// Kernel 0: zero the accumulator region (must be re-zeroed every call).
// ---------------------------------------------------------------------------
__global__ void tasp_zero_kernel(float* __restrict__ p, int n)
{
    int i = blockIdx.x * blockDim.x + threadIdx.x;
    if (i < n) p[i] = 0.f;
}

extern "C" void kernel_launch(void* const* d_in, const int* in_sizes, int n_in,
                              void* d_out, int out_size, void* d_ws, size_t ws_size,
                              hipStream_t stream)
{
    const float* x  = (const float*)d_in[0];   // [B, T, D]
    const float* W1 = (const float*)d_in[1];   // [D, A]
    const float* b1 = (const float*)d_in[2];   // [A]
    const float* W2 = (const float*)d_in[3];   // [A, 1]
    const float* b2 = (const float*)d_in[4];   // [1]
    float* out = (float*)d_out;                // [B, 2D]

    // workspace layout (floats)
    float* ws       = (float*)d_ws;
    float* scores   = ws;                       // B*T = 131072
    float* weights  = ws + (size_t)BB * TT;     // B*T = 131072
    float* acc      = weights + (size_t)BB * TT;// B*2*D = 65536

    const int accN = BB * 2 * DD;
    tasp_zero_kernel<<<(accN + 255) / 256, 256, 0, stream>>>(acc, accN);

    // 16 rows/wave * 8 waves/block = 128 rows per block
    const int nblocks = (BB * TT) / 128;        // 1024
    tasp_score_kernel<<<nblocks, 256, 0, stream>>>(x, W1, b1, W2, b2, scores);

    tasp_softmax_kernel<<<BB, 256, 0, stream>>>(scores, weights);

    dim3 sgrid(BB, 16);
    tasp_stats_kernel<<<sgrid, 128, 0, stream>>>(x, weights, acc);

    tasp_finalize_kernel<<<BB, 512, 0, stream>>>(acc, out);
}
